// MultiHeadAttention_5789615915542
// MI455X (gfx1250) — matrix operand order repaired and checked by reading, not compiled, not run on hardware
//
#include <hip/hip_runtime.h>

// ---------------------------------------------------------------------------
// MHA forward for MI455X (gfx1250), wave32, WMMA f16 16x16x32.
// B=2, S=2048, D=1024, H=16, K=V=64.  All GEMM dims hardcoded (4096x1024x1024).
// q/k/v intermediates stored f16 (V pre-transposed) so attention staging is a
// raw byte copy -> CDNA5 async global->LDS path (ASYNCcnt).
// ---------------------------------------------------------------------------

typedef __attribute__((ext_vector_type(16))) _Float16 v16h;
typedef __attribute__((ext_vector_type(8)))  _Float16 v8h;
typedef __attribute__((ext_vector_type(4)))  _Float16 v4h;
typedef __attribute__((ext_vector_type(8)))  float    v8f;
typedef __attribute__((ext_vector_type(4)))  int      v4i;

#define BB   2
#define SS   2048
#define DD   1024
#define HH   16
#define HD   64

#define GLOBAL_AS __attribute__((address_space(1)))
#define LDS_AS    __attribute__((address_space(3)))

static __device__ __forceinline__ v8f wmma_f16(v16h a, v16h b, v8f c) {
  return __builtin_amdgcn_wmma_f32_16x16x32_f16(
      false, a, false, b, (short)0, c, false, false);
}

// 16B async copy global->LDS (gfx1250 GLOBAL_LOAD_ASYNC_TO_LDS_B128).
static __device__ __forceinline__ void async_copy_b128(const void* gsrc, void* ldst) {
#if __has_builtin(__builtin_amdgcn_global_load_async_to_lds_b128)
  __builtin_amdgcn_global_load_async_to_lds_b128(
      (GLOBAL_AS v4i*)gsrc, (LDS_AS v4i*)ldst, 0, 0);
#else
  *(v8h*)ldst = *(const v8h*)gsrc;
#endif
}

static __device__ __forceinline__ void wait_async0() {
#if __has_builtin(__builtin_amdgcn_s_wait_asynccnt)
  __builtin_amdgcn_s_wait_asynccnt(0);
#else
  asm volatile("s_wait_asynccnt 0" ::: "memory");
#endif
}

// A-fragment (16x32 f16) from row-major LDS tile [m][k]: two b128 chunks.
static __device__ __forceinline__ v16h load_frag_a(const _Float16* lds,
                                                   int row0, int ld, int kbase) {
  const int lane = threadIdx.x & 31;
  const int m    = lane & 15;
  const int koff = (lane >> 4) << 3;
  const _Float16* p = lds + (row0 + m) * ld + kbase + koff;
  const v8h lo = *(const v8h*)(p);
  const v8h hi = *(const v8h*)(p + 16);
  return __builtin_shufflevector(lo, hi, 0, 1, 2, 3, 4, 5, 6, 7,
                                 8, 9, 10, 11, 12, 13, 14, 15);
}

// B-fragment (32x16 f16) from transposed LDS tile [n][k]: two b128 chunks.
static __device__ __forceinline__ v16h load_frag_bt(const _Float16* lds,
                                                    int col0, int ld, int kbase) {
  const int lane = threadIdx.x & 31;
  const int n    = lane & 15;
  const int koff = (lane >> 4) << 3;
  const _Float16* p = lds + (col0 + n) * ld + kbase + koff;
  const v8h lo = *(const v8h*)(p);
  const v8h hi = *(const v8h*)(p + 16);
  return __builtin_shufflevector(lo, hi, 0, 1, 2, 3, 4, 5, 6, 7,
                                 8, 9, 10, 11, 12, 13, 14, 15);
}

static __device__ __forceinline__ v4h cvt4(float4 f) {
  v4h h;
  h[0] = (_Float16)f.x; h[1] = (_Float16)f.y;
  h[2] = (_Float16)f.z; h[3] = (_Float16)f.w;
  return h;
}

// ---------------------------------------------------------------------------
// GEMM: C = (A[4096,1024] * W[1024,1024] + bias) * scale
// OMODE 0: f32 row-major [M,N]        (final output projection)
// OMODE 1: f16 row-major [M,N]        (q, k projections)
// OMODE 2: f16 transposed [b][col][t] (v projection -> [B,H*64v,S])
// ---------------------------------------------------------------------------
template <int OMODE>
__global__ __launch_bounds__(128)
void mha_gemm_kernel(const float* __restrict__ A,
                     const float* __restrict__ W,
                     const float* __restrict__ bias,
                     void* __restrict__ Cout,
                     float scale) {
  constexpr int NN = 1024, KD = 1024;
  __shared__ __align__(16) _Float16 sA [64 * 32];  // [m][k]
  __shared__ __align__(16) _Float16 sBt[64 * 32];  // [n][k]

  const int tid  = threadIdx.x;
  const int lane = tid & 31;
  const int wave = tid >> 5;
  const int wm   = (wave >> 1) * 32;
  const int wn   = (wave & 1) * 32;
  const int row0 = blockIdx.y * 64;
  const int col0 = blockIdx.x * 64;

  v8f acc[2][2] = {};

  for (int kk = 0; kk < KD; kk += 32) {
    if (kk + 32 < KD) {
      __builtin_prefetch(&A[(size_t)(row0 + (tid >> 1)) * KD + kk + 32], 0, 3);
      __builtin_prefetch(&W[(size_t)(kk + 32 + (tid >> 4)) * NN + col0 + ((tid & 15) << 2)], 0, 3);
    }
#pragma unroll
    for (int i = 0; i < 4; ++i) {
      const int e = tid + i * 128;
      const int r = e >> 3, q = (e & 7) << 2;
      const float4 a4 = *(const float4*)&A[(size_t)(row0 + r) * KD + kk + q];
      *(v4h*)&sA[r * 32 + q] = cvt4(a4);
    }
#pragma unroll
    for (int i = 0; i < 4; ++i) {
      const int e = tid + i * 128;
      const int n = e & 63, kq = (e >> 6) << 2;
      v4h h4;
#pragma unroll
      for (int j = 0; j < 4; ++j)
        h4[j] = (_Float16)W[(size_t)(kk + kq + j) * NN + col0 + n];
      *(v4h*)&sBt[n * 32 + kq] = h4;
    }
    __syncthreads();

    const v16h a0 = load_frag_a (sA,  wm,      32, 0);
    const v16h a1 = load_frag_a (sA,  wm + 16, 32, 0);
    const v16h b0 = load_frag_bt(sBt, wn,      32, 0);
    const v16h b1 = load_frag_bt(sBt, wn + 16, 32, 0);
    acc[0][0] = wmma_f16(a0, b0, acc[0][0]);
    acc[0][1] = wmma_f16(a0, b1, acc[0][1]);
    acc[1][0] = wmma_f16(a1, b0, acc[1][0]);
    acc[1][1] = wmma_f16(a1, b1, acc[1][1]);
    __syncthreads();
  }

  const int mbase = (lane >> 4) << 3;
  const int nn    = lane & 15;
#pragma unroll
  for (int i = 0; i < 2; ++i)
#pragma unroll
    for (int j = 0; j < 2; ++j)
#pragma unroll
      for (int r = 0; r < 8; ++r) {
        const int row = row0 + wm + i * 16 + mbase + r;
        const int col = col0 + wn + j * 16 + nn;
        const float val = (acc[i][j][r] + bias[col]) * scale;
        if constexpr (OMODE == 0) {
          ((float*)Cout)[(size_t)row * NN + col] = val;
        } else if constexpr (OMODE == 1) {
          ((_Float16*)Cout)[(size_t)row * NN + col] = (_Float16)val;
        } else {
          const int bI = row >> 11, t = row & 2047;   // row = b*S + t
          ((_Float16*)Cout)[((size_t)bI * 1024 + col) * SS + t] = (_Float16)val;
        }
      }
}

// ---------------------------------------------------------------------------
// Flash attention: grid (S/64, B*H), 128 threads (4 waves x 16 query rows).
// Qh/Kh: f16 [B,S,H,64] (Q carries 1/sqrt(K)).  Vt: f16 [B,H*64v,S].
// All tiles staged via async global->LDS b128 copies.
// ---------------------------------------------------------------------------
__global__ __launch_bounds__(128)
void mha_flash_attn_kernel(const _Float16* __restrict__ Qh,
                           const _Float16* __restrict__ Kh,
                           const _Float16* __restrict__ Vt,
                           const float* __restrict__ maskg,
                           float* __restrict__ Og) {
  __shared__ __align__(16) _Float16 sQ [64 * 64];    // [qrow][k]   (A operand)
  __shared__ __align__(16) _Float16 sK [64 * 64];    // [keyrow][k] (B operand, [n][k])
  __shared__ __align__(16) _Float16 sVt[64 * 64];    // [v][s]      (B operand, [n][k])
  __shared__ __align__(16) _Float16 sP [4][16 * 64]; // per-wave P tile (A operand)

  const int tid  = threadIdx.x;
  const int lane = tid & 31;
  const int wave = tid >> 5;
  const int bh   = blockIdx.y;
  const int b    = bh >> 4;
  const int h    = bh & 15;
  const int q0   = blockIdx.x * 64;
  const int wm   = wave * 16;

  // Stage Q block: 64 rows x 128B, async b128 (4 chunks/thread).
#pragma unroll
  for (int i = 0; i < 4; ++i) {
    const int e = tid + i * 128;            // 512 chunks of 8 halves
    const int r = e >> 3, c8 = (e & 7) << 3;
    async_copy_b128(&Qh[(((size_t)b * SS + q0 + r) * HH + h) * HD + c8],
                    &sQ[r * 64 + c8]);
  }
  wait_async0();
  __syncthreads();

  const v16h aq0 = load_frag_a(sQ, wm, 64, 0);
  const v16h aq1 = load_frag_a(sQ, wm, 64, 32);

  const int mbase = (lane >> 4) << 3;
  const int nn    = lane & 15;

  float m_i[8], l_i[8];
#pragma unroll
  for (int r = 0; r < 8; ++r) { m_i[r] = -1e30f; l_i[r] = 0.0f; }
  v8f o[4] = {};

  for (int sb = 0; sb < SS; sb += 64) {
    // K block rows (already the [n][k] layout B fragments want).
#pragma unroll
    for (int i = 0; i < 4; ++i) {
      const int e = tid + i * 128;
      const int r = e >> 3, c8 = (e & 7) << 3;
      async_copy_b128(&Kh[(((size_t)b * SS + sb + r) * HH + h) * HD + c8],
                      &sK[r * 64 + c8]);
    }
    // V^T rows: Vt[b][h*64+v][s] -> sVt[v][s].
#pragma unroll
    for (int i = 0; i < 4; ++i) {
      const int e = tid + i * 128;
      const int r = e >> 3, c8 = (e & 7) << 3;
      async_copy_b128(&Vt[((size_t)b * 1024 + h * HD + r) * SS + sb + c8],
                      &sVt[r * 64 + c8]);
    }
    wait_async0();
    __syncthreads();

    // scores: wave's 16 rows x 64 key cols
    v8f sc[4] = {};
#pragma unroll
    for (int t = 0; t < 4; ++t) {
      const v16h bk0 = load_frag_bt(sK, t * 16, 64, 0);
      const v16h bk1 = load_frag_bt(sK, t * 16, 64, 32);
      sc[t] = wmma_f16(aq0, bk0, sc[t]);
      sc[t] = wmma_f16(aq1, bk1, sc[t]);
    }

    // additive mask: (1 - m) * -10000
#pragma unroll
    for (int t = 0; t < 4; ++t)
#pragma unroll
      for (int r = 0; r < 8; ++r) {
        const int trow = q0 + wm + mbase + r;
        const int scol = sb + t * 16 + nn;
        const float mv = maskg[((size_t)b * SS + trow) * SS + scol];
        sc[t][r] += (1.0f - mv) * -10000.0f;
      }

    // online softmax (row state replicated across each 16-lane half)
#pragma unroll
    for (int r = 0; r < 8; ++r) {
      float mx = fmaxf(fmaxf(sc[0][r], sc[1][r]), fmaxf(sc[2][r], sc[3][r]));
#pragma unroll
      for (int off = 8; off >= 1; off >>= 1)
        mx = fmaxf(mx, __shfl_xor(mx, off, 32));
      const float mnew = fmaxf(m_i[r], mx);
      const float corr = __expf(m_i[r] - mnew);
      float psum = 0.0f;
#pragma unroll
      for (int t = 0; t < 4; ++t) {
        const float p = __expf(sc[t][r] - mnew);
        sc[t][r] = p;
        psum += p;
      }
#pragma unroll
      for (int off = 8; off >= 1; off >>= 1)
        psum += __shfl_xor(psum, off, 32);
      l_i[r] = l_i[r] * corr + psum;
      m_i[r] = mnew;
#pragma unroll
      for (int t = 0; t < 4; ++t) o[t][r] *= corr;
    }

    // C-layout P -> per-wave LDS tile (row-major A operand)
#pragma unroll
    for (int t = 0; t < 4; ++t)
#pragma unroll
      for (int r = 0; r < 8; ++r)
        sP[wave][(mbase + r) * 64 + t * 16 + nn] = (_Float16)sc[t][r];
    asm volatile("s_wait_dscnt 0" ::: "memory");   // wave-local LDS RAW

    const v16h ap0 = load_frag_a(&sP[wave][0], 0, 64, 0);
    const v16h ap1 = load_frag_a(&sP[wave][0], 0, 64, 32);
#pragma unroll
    for (int t = 0; t < 4; ++t) {
      const v16h bv0 = load_frag_bt(sVt, t * 16, 64, 0);
      const v16h bv1 = load_frag_bt(sVt, t * 16, 64, 32);
      o[t] = wmma_f16(ap0, bv0, o[t]);
      o[t] = wmma_f16(ap1, bv1, o[t]);
    }
    __syncthreads();
  }

  // normalize and store attn [B,S,H,64] f32 (A operand of final GEMM)
#pragma unroll
  for (int t = 0; t < 4; ++t)
#pragma unroll
    for (int r = 0; r < 8; ++r) {
      const int row = q0 + wm + mbase + r;
      const int v   = t * 16 + nn;
      Og[(((size_t)b * SS + row) * HH + h) * HD + v] = o[t][r] / l_i[r];
    }
}

// ---------------------------------------------------------------------------
// Host-side orchestration.
// Input order: query, value, key, attention_mask, Wq, bq, Wk, bk, Wv, bv, Wo, bo
// ---------------------------------------------------------------------------
extern "C" void kernel_launch(void* const* d_in, const int* in_sizes, int n_in,
                              void* d_out, int out_size, void* d_ws, size_t ws_size,
                              hipStream_t stream) {
  const float* query = (const float*)d_in[0];
  const float* value = (const float*)d_in[1];
  const float* key_i = (const float*)d_in[2];
  const float* mask  = (const float*)d_in[3];
  const float* Wq    = (const float*)d_in[4];
  const float* bq    = (const float*)d_in[5];
  const float* Wk    = (const float*)d_in[6];
  const float* bk    = (const float*)d_in[7];
  const float* Wv    = (const float*)d_in[8];
  const float* bv    = (const float*)d_in[9];
  const float* Wo    = (const float*)d_in[10];
  const float* bo    = (const float*)d_in[11];
  float* out = (float*)d_out;

  const size_t SZ = (size_t)BB * SS * DD;       // 4M elements
  _Float16* qbuf = (_Float16*)d_ws;             // f16 [B,S,H,64]
  _Float16* kbuf = qbuf + SZ;                   // f16 [B,S,H,64]
  _Float16* vtbuf = kbuf + SZ;                  // f16 [B,H*64,S]
  float*    abuf  = (float*)(vtbuf + SZ);       // f32 [B,S,H,64]

  const dim3 ggrid(16, 64);                     // N/64, M/64
  const dim3 gblk(128);
  const float qscale = 0.125f;                  // 1/sqrt(64)

  mha_gemm_kernel<1><<<ggrid, gblk, 0, stream>>>(query, Wq, bq, qbuf, qscale);
  mha_gemm_kernel<1><<<ggrid, gblk, 0, stream>>>(key_i, Wk, bk, kbuf, 1.0f);
  mha_gemm_kernel<2><<<ggrid, gblk, 0, stream>>>(value, Wv, bv, vtbuf, 1.0f);

  const dim3 agrid(SS / 64, BB * HH);           // (32, 32)
  mha_flash_attn_kernel<<<agrid, gblk, 0, stream>>>(qbuf, kbuf, vtbuf, mask, abuf);

  mha_gemm_kernel<0><<<ggrid, gblk, 0, stream>>>(abuf, Wo, bo, out, 1.0f);
}
